// W3A4Linear_72052371357903
// MI455X (gfx1250) — matrix operand order repaired and checked by reading, not compiled, not run on hardware
//
#include <hip/hip_runtime.h>
#include <hip/hip_bf16.h>

// ---------------- problem constants (from reference) ----------------
#define IN_F   1152
#define OUT_F  4608
#define RANK   32
#define HADB   128
#define M_ROWS 16384          // 4 * 4096
#define KCH    (IN_F / 32)    // 36 k-chunks of 32 for WMMA f16

typedef __attribute__((ext_vector_type(16))) _Float16 v16h;
typedef __attribute__((ext_vector_type(4)))  _Float16 v4h;
typedef __attribute__((ext_vector_type(8)))  float    v8f;

// ---------------- WMMA wrapper (CDNA5, wave32) ----------------
__device__ __forceinline__ v8f wmma_f16(v16h a, v16h b, v8f c) {
  // D = A(16x32 f16) * B(32x16 f16) + C(16x16 f32)
  return __builtin_amdgcn_wmma_f32_16x16x32_f16(
      /*neg_a=*/false, a, /*neg_b=*/false, b,
      /*c_mod=*/(short)0, c, /*reuse_a=*/false, /*reuse_b=*/false);
}

// ---------------- fragment loaders ----------------
// ISA 16-bit A-matrix 16x32 layout: lane holds row m=lane&15;
// kb = (lane>>4)*8; halves 0..7 = K kb..kb+7, halves 8..15 = K kb+16..kb+23.
// Source is row-major f16 [M][ld]; both contiguous runs are 16B -> 2x b128.
__device__ __forceinline__ v16h load_a_frag(const _Float16* __restrict__ p, int ld) {
  int lane = threadIdx.x & 31;
  const _Float16* row = p + (size_t)(lane & 15) * ld + ((lane >> 4) << 3);
  v16h f;
  ((float4*)&f)[0] = *(const float4*)(row);
  ((float4*)&f)[1] = *(const float4*)(row + 16);
  return f;
}

// ISA 16-bit B-matrix 32x16 layout: lane holds col n=lane&15;
// kb = (lane>>4)*8; halves 0..7 = K kb..kb+7, halves 8..15 = K kb+16..kb+23.
// Source stored K-contiguous per column: row-major [N][ld] (i.e. B[k][n]=src[n][k]).
__device__ __forceinline__ v16h load_b_frag(const _Float16* __restrict__ p, int ld) {
  int lane = threadIdx.x & 31;
  const _Float16* row = p + (size_t)(lane & 15) * ld + ((lane >> 4) << 3);
  v16h f;
  ((float4*)&f)[0] = *(const float4*)(row);
  ((float4*)&f)[1] = *(const float4*)(row + 16);
  return f;
}

// C/D 16x16 f32 layout: component g sits at row m = g + 8*(lane>>4), col = lane&15.

// ---------------- kernel 0: f32 -> f16 convert (W, H, lora_a, lora_b) ------
__global__ __launch_bounds__(256) void cvt_f32_f16(const float* __restrict__ src,
                                                   _Float16* __restrict__ dst,
                                                   int n4) {
  int i = blockIdx.x * blockDim.x + threadIdx.x;
  if (i < n4) {
    float4 v = ((const float4*)src)[i];
    v4h h;
    h[0] = (_Float16)v.x; h[1] = (_Float16)v.y;
    h[2] = (_Float16)v.z; h[3] = (_Float16)v.w;
    ((v4h*)dst)[i] = h;
  }
}

// ---------------- kernel 1: fused preprocess --------------------------------
// One wave per workgroup handles 16 rows of x:
//   Hadamard rotate (WMMA, H symmetric so B-frag = row-major H),
//   * smooth_scale, register group-quant (group==N-tile), write x_q f16,
//   then r = x_smooth @ lora_a^T (WMMA over K=1152) -> r16[M][32].
__global__ __launch_bounds__(32) void preprocess_kernel(
    const float* __restrict__ x, const _Float16* __restrict__ H16,
    const float* __restrict__ smooth, const _Float16* __restrict__ la16,
    _Float16* __restrict__ xq16, _Float16* __restrict__ r16) {
  __shared__ __align__(16) _Float16 xblk[16 * HADB];   // one Hadamard block, f16
  __shared__ __align__(16) _Float16 xs16[16 * IN_F];   // smoothed row tile, f16

  const int lane = threadIdx.x;
  const int row0 = blockIdx.x * 16;
  const int half = (lane >> 4);          // 0 or 1
  const int nloc = lane & 15;

  for (int b = 0; b < IN_F / HADB; ++b) {
    // stage 16x128 f32 -> f16 into LDS (each lane: 16 float4 loads)
    for (int i = lane; i < 16 * HADB / 4; i += 32) {
      int r = i >> 5, c4 = i & 31;
      float4 v = *(const float4*)(x + (size_t)(row0 + r) * IN_F + b * HADB + c4 * 4);
      _Float16* d = &xblk[r * HADB + c4 * 4];
      d[0] = (_Float16)v.x; d[1] = (_Float16)v.y;
      d[2] = (_Float16)v.z; d[3] = (_Float16)v.w;
    }
    __syncthreads();

    for (int n = 0; n < HADB / 16; ++n) {
      v8f acc = {};
      #pragma unroll
      for (int kc = 0; kc < HADB / 32; ++kc) {
        v16h a  = load_a_frag(xblk + kc * 32, HADB);
        v16h bb = load_b_frag(H16 + (size_t)(n * 16) * HADB + kc * 32, HADB);
        acc = wmma_f16(a, bb, acc);
      }
      const int col = b * HADB + n * 16 + nloc;
      const float s = smooth[col];
      #pragma unroll
      for (int g = 0; g < 8; ++g) acc[g] *= s;

      // keep pre-quant value for the LoRA path
      #pragma unroll
      for (int g = 0; g < 8; ++g)
        xs16[(g + half * 8) * IN_F + col] = (_Float16)acc[g];

      // fake-INT4 quantize: group of 16 channels == this N-tile; amax is a
      // 16-lane half-wave reduction over one accumulator component.
      #pragma unroll
      for (int g = 0; g < 8; ++g) {
        float v = acc[g];
        float am = fabsf(v);
        am = fmaxf(am, __shfl_xor(am, 1, 16));
        am = fmaxf(am, __shfl_xor(am, 2, 16));
        am = fmaxf(am, __shfl_xor(am, 4, 16));
        am = fmaxf(am, __shfl_xor(am, 8, 16));
        float scale = fmaxf(am, 1e-12f) * (1.0f / 7.0f);
        float q = rintf(v / scale);
        q = fminf(7.0f, fmaxf(-7.0f, q)) * scale;
        xq16[(size_t)(row0 + g + half * 8) * IN_F + col] = (_Float16)q;
      }
    }
    __syncthreads();
  }

  // r = x_smooth @ lora_a^T   (16 x 32), WMMA over K=1152
  #pragma unroll
  for (int nt = 0; nt < RANK / 16; ++nt) {
    v8f acc = {};
    for (int kc = 0; kc < KCH; ++kc) {
      v16h a  = load_a_frag(xs16 + kc * 32, IN_F);
      v16h bb = load_b_frag(la16 + (size_t)(nt * 16) * IN_F + kc * 32, IN_F);
      acc = wmma_f16(a, bb, acc);
    }
    #pragma unroll
    for (int g = 0; g < 8; ++g)
      r16[(size_t)(row0 + g + half * 8) * RANK + nt * 16 + nloc] = (_Float16)acc[g];
  }
}

// ---------------- kernel 2: main WMMA GEMM + LoRA-B + bias ------------------
// Each wave owns a 32x64 output tile (2 A-frags x 4 B-frags -> 8 WMMAs per
// k-chunk); B fragments are shared across both row-halves, halving L2 weight
// traffic vs a 16-row tile. A/B stream straight from global: the whole f16
// working set (~48 MB) is L2-resident on the 192 MB L2.
__global__ __launch_bounds__(256) void gemm_kernel(
    const _Float16* __restrict__ xq16, const _Float16* __restrict__ W16,
    const _Float16* __restrict__ r16,  const _Float16* __restrict__ lb16,
    const float* __restrict__ bias, float* __restrict__ out) {
  const int lane = threadIdx.x & 31;
  const int wave = threadIdx.x >> 5;
  const int tile = blockIdx.x * 8 + wave;
  const int NT64 = OUT_F / 64;                 // 72
  const int mt = tile / NT64, nt = tile % NT64;
  if (mt >= M_ROWS / 32) return;               // wave-uniform (EXEC stays all-1)
  const int row0 = mt * 32, col0 = nt * 64;
  const int half = (lane >> 4), nloc = lane & 15;

  v8f acc[2][4] = {};
  const _Float16* ap0 = xq16 + (size_t)row0 * IN_F;
  const _Float16* ap1 = ap0 + (size_t)16 * IN_F;

  for (int kc = 0; kc < KCH; ++kc) {
    if (kc + 4 < KCH) {
      __builtin_prefetch(ap0 + (size_t)nloc * IN_F + (kc + 4) * 32, 0, 0);
      __builtin_prefetch(ap1 + (size_t)nloc * IN_F + (kc + 4) * 32, 0, 0);
    }
    v16h a0 = load_a_frag(ap0 + kc * 32, IN_F);
    v16h a1 = load_a_frag(ap1 + kc * 32, IN_F);
    #pragma unroll
    for (int j = 0; j < 4; ++j) {
      v16h bb = load_b_frag(W16 + (size_t)(col0 + j * 16) * IN_F + kc * 32, IN_F);
      acc[0][j] = wmma_f16(a0, bb, acc[0][j]);
      acc[1][j] = wmma_f16(a1, bb, acc[1][j]);
    }
  }

  // LoRA: one extra K=32 WMMA per (row-half, N-tile)
  v16h ra0 = load_a_frag(r16 + (size_t)row0 * RANK, RANK);
  v16h ra1 = load_a_frag(r16 + (size_t)(row0 + 16) * RANK, RANK);
  #pragma unroll
  for (int j = 0; j < 4; ++j) {
    v16h lb = load_b_frag(lb16 + (size_t)(col0 + j * 16) * RANK, RANK);
    acc[0][j] = wmma_f16(ra0, lb, acc[0][j]);
    acc[1][j] = wmma_f16(ra1, lb, acc[1][j]);
  }

  // epilogue: + bias, f32 store (16 lanes/row -> 64B coalesced segments)
  #pragma unroll
  for (int j = 0; j < 4; ++j) {
    const int col = col0 + j * 16 + nloc;
    const float bv = bias[col];
    #pragma unroll
    for (int h = 0; h < 2; ++h) {
      #pragma unroll
      for (int g = 0; g < 8; ++g)
        out[(size_t)(row0 + h * 16 + g + half * 8) * OUT_F + col] = acc[h][j][g] + bv;
    }
  }
}

// ---------------- host launcher ---------------------------------------------
extern "C" void kernel_launch(void* const* d_in, const int* in_sizes, int n_in,
                              void* d_out, int out_size, void* d_ws, size_t ws_size,
                              hipStream_t stream) {
  const float* x      = (const float*)d_in[0];  // [4,4096,1152]
  const float* Hb     = (const float*)d_in[1];  // [128,128]
  const float* smooth = (const float*)d_in[2];  // [1152]
  const float* w      = (const float*)d_in[3];  // [4608,1152]
  const float* la     = (const float*)d_in[4];  // [32,1152]
  const float* lb     = (const float*)d_in[5];  // [4608,32]
  const float* bias   = (const float*)d_in[6];  // [4608]
  float* out = (float*)d_out;

  // workspace partition (256B aligned slices)
  char* wsp = (char*)d_ws;
  size_t off = 0;
  auto take = [&](size_t bytes) -> char* {
    char* p = wsp + off;
    off += (bytes + 255) & ~(size_t)255;
    return p;
  };
  _Float16* W16  = (_Float16*)take((size_t)OUT_F * IN_F * 2);
  _Float16* H16  = (_Float16*)take((size_t)HADB * HADB * 2);
  _Float16* la16 = (_Float16*)take((size_t)RANK * IN_F * 2);
  _Float16* lb16 = (_Float16*)take((size_t)OUT_F * RANK * 2);
  _Float16* xq16 = (_Float16*)take((size_t)M_ROWS * IN_F * 2);
  _Float16* r16  = (_Float16*)take((size_t)M_ROWS * RANK * 2);
  (void)ws_size; (void)n_in; (void)in_sizes; (void)out_size;

  // 0) precision convert the static operands
  auto cvt = [&](const float* s, _Float16* d, size_t n) {
    int n4 = (int)(n / 4);
    cvt_f32_f16<<<(n4 + 255) / 256, 256, 0, stream>>>(s, d, n4);
  };
  cvt(w,  W16,  (size_t)OUT_F * IN_F);
  cvt(Hb, H16,  (size_t)HADB * HADB);   // H symmetric: row-major == col-major
  cvt(la, la16, (size_t)RANK * IN_F);
  cvt(lb, lb16, (size_t)OUT_F * RANK);

  // 1) fused Hadamard + smooth + INT4 fake-quant + LoRA-A projection
  preprocess_kernel<<<M_ROWS / 16, 32, 0, stream>>>(x, H16, smooth, la16, xq16, r16);

  // 2) main GEMM + LoRA-B + bias (32x64 per wave)
  const int wave_tiles = (M_ROWS / 32) * (OUT_F / 64);   // 36864
  gemm_kernel<<<wave_tiles / 8, 256, 0, stream>>>(xq16, W16, r16, lb16, bias, out);
}
